// MambaModel_50036368998656
// MI455X (gfx1250) — matrix operand order repaired and checked by reading, not compiled, not run on hardware
//
#include <hip/hip_runtime.h>
#include <cstdint>
#include <cstddef>

// ---------------------------------------------------------------------------
// Model dimensions (match reference setup_inputs)
// ---------------------------------------------------------------------------
constexpr int LSEQ   = 1024;
constexpr int BATCH  = 4;
constexpr int FIN    = 128;
constexpr int DMODEL = 1024;
constexpr int DINNER = 2048;
constexpr int NSTATE = 16;
constexpr int KCONV  = 4;
constexpr int RRANK  = 64;
constexpr int HDEC   = 2048;
constexpr int NOUTC  = 10;
constexpr int NLAYER = 4;
constexpr int MROWS  = BATCH * LSEQ;          // 4096 token rows, layout (B, L, ...)
constexpr int XPROJC = RRANK + 2 * NSTATE;    // 96

typedef __attribute__((ext_vector_type(16))) __bf16 v16bf;
typedef __attribute__((ext_vector_type(8)))  __bf16 v8bf;
typedef __attribute__((ext_vector_type(8)))  float  v8f;

__device__ __forceinline__ float sigmoidf_(float x) { return 1.f / (1.f + __expf(-x)); }
__device__ __forceinline__ float siluf_(float x)    { return x * sigmoidf_(x); }
__device__ __forceinline__ float softplusf_(float x){ return (x > 20.f) ? x : log1pf(__expf(x)); }
__device__ __forceinline__ float geluf_(float x)    { return 0.5f * x * (1.f + erff(x * 0.70710678118654752f)); }

// ---------------------------------------------------------------------------
// CDNA5 async global->LDS (ASYNCcnt-tracked, no VGPR staging).
// LDS operand is the low 32 bits of the generic shared pointer.
// ---------------------------------------------------------------------------
__device__ __forceinline__ void async_ld_b128(void* lds_ptr, const void* gptr) {
    asm volatile("global_load_async_to_lds_b128 %0, %1, off"
                 :: "v"((uint32_t)(uintptr_t)lds_ptr),
                    "v"((unsigned long long)(uintptr_t)gptr)
                 : "memory");
}
#define WAIT_ASYNC(n) asm volatile("s_wait_asynccnt " #n ::: "memory")

// ---------------------------------------------------------------------------
// fp32 -> bf16 conversion (weights)
// ---------------------------------------------------------------------------
__global__ void f32_to_bf16_kernel(const float* __restrict__ in, __bf16* __restrict__ out, size_t n) {
    size_t i = (size_t)blockIdx.x * blockDim.x + threadIdx.x;
    if (i < n) out[i] = (__bf16)in[i];
}

// ---------------------------------------------------------------------------
// Encoder: h[b*L+l, d] = x_src[l,b,:] @ enc_w[d,:] + enc_b[d] (+ y path if l<sep)
// ---------------------------------------------------------------------------
__global__ __launch_bounds__(256) void encoder_kernel(
    const float* __restrict__ x_src, const float* __restrict__ y_src,
    const float* __restrict__ enc_w, const float* __restrict__ enc_b,
    const float* __restrict__ yenc_w, const float* __restrict__ yenc_b,
    const int* __restrict__ sep_p, float* __restrict__ h)
{
    __shared__ float sx[FIN];
    const int row = blockIdx.x;            // b*L + l
    const int b = row >> 10;               // /LSEQ
    const int l = row & (LSEQ - 1);
    const int tid = threadIdx.x;
    if (tid < FIN) sx[tid] = x_src[((size_t)l * BATCH + b) * FIN + tid];
    __syncthreads();
    const int sep = *sep_p;
    const float yv = y_src[l * BATCH + b];
    const bool addy = (l < sep);
    for (int j = 0; j < DMODEL / 256; ++j) {
        const int d = tid + j * 256;
        const float4* w4 = (const float4*)(enc_w + (size_t)d * FIN);
        float acc = enc_b[d];
#pragma unroll 8
        for (int f = 0; f < FIN / 4; ++f) {
            float4 wv = w4[f];
            acc += wv.x * sx[f * 4 + 0] + wv.y * sx[f * 4 + 1] +
                   wv.z * sx[f * 4 + 2] + wv.w * sx[f * 4 + 3];
        }
        if (addy) acc += yv * yenc_w[d] + yenc_b[d];
        h[(size_t)row * DMODEL + d] = acc;
    }
}

// ---------------------------------------------------------------------------
// Residual add + LayerNorm over D=1024, emits bf16 normalized activations
// ---------------------------------------------------------------------------
__global__ __launch_bounds__(256) void ln_kernel(
    const float* __restrict__ hin, float* __restrict__ res,
    const float* __restrict__ w, const float* __restrict__ bias,
    __bf16* __restrict__ out, int first)
{
    __shared__ float red[256], red2[256], stats[2];
    const size_t row = blockIdx.x;
    const int tid = threadIdx.x;
    float v[DMODEL / 256];
    float s = 0.f, s2 = 0.f;
#pragma unroll
    for (int j = 0; j < DMODEL / 256; ++j) {
        const int d = tid + j * 256;
        float r = hin[row * DMODEL + d];
        if (!first) r += res[row * DMODEL + d];
        res[row * DMODEL + d] = r;
        v[j] = r; s += r; s2 += r * r;
    }
    red[tid] = s; red2[tid] = s2;
    __syncthreads();
    for (int st = 128; st > 0; st >>= 1) {
        if (tid < st) { red[tid] += red[tid + st]; red2[tid] += red2[tid + st]; }
        __syncthreads();
    }
    if (tid == 0) {
        float m = red[0] * (1.f / DMODEL);
        float var = red2[0] * (1.f / DMODEL) - m * m;
        stats[0] = m; stats[1] = rsqrtf(var + 1e-5f);
    }
    __syncthreads();
    const float m = stats[0], inv = stats[1];
#pragma unroll
    for (int j = 0; j < DMODEL / 256; ++j) {
        const int d = tid + j * 256;
        out[row * DMODEL + d] = (__bf16)((v[j] - m) * inv * w[d] + bias[d]);
    }
}

// ---------------------------------------------------------------------------
// Generic NT GEMM with bf16 WMMA:  C[M,N] = A[M,K] * W[N,K]^T
// A row-major (lda), W row-major rows of K (ldw=K). Block tile 256x128x64,
// 8 waves in a 4x2 grid of 64x64 wave tiles (16 WMMA : 16 ds_load_b128 per
// kk-step, LDS-bandwidth balanced). Double-buffered LDS fed by
// GLOBAL_LOAD_ASYNC_TO_LDS_B128 (ASYNCcnt). M is a multiple of 256; W
// over-reads past N land in adjacent workspace (harmless: epilogue never
// stores gn >= N).
// EPI: 0=f32 out, 1=bias+softplus f32, 2=bias+gelu bf16, 3=f32+bf16 dual
// ---------------------------------------------------------------------------
constexpr int BM = 256, BN = 128, BKT = 64, LDSS = 72; // 8-elem pad per LDS row

template <int EPI>
__global__ __launch_bounds__(256) void gemm_nt_kernel(
    const __bf16* __restrict__ A, int lda,
    const __bf16* __restrict__ W,
    float* __restrict__ Cf, __bf16* __restrict__ Cb, int ldc,
    const float* __restrict__ bias,
    int M, int N, int K)
{
    __shared__ __bf16 sA[2 * BM * LDSS];
    __shared__ __bf16 sB[2 * BN * LDSS];
    const int tid  = threadIdx.x;
    const int lane = tid & 31;
    const int wid  = tid >> 5;       // 0..7
    const int wr   = wid >> 1;       // 0..3 -> 64-row band
    const int wc   = wid & 1;        // 0..1 -> 64-col band
    const int m0   = blockIdx.y * BM;
    const int n0   = blockIdx.x * BN;

    v8f acc[4][4];
#pragma unroll
    for (int i = 0; i < 4; ++i)
#pragma unroll
        for (int j = 0; j < 4; ++j)
#pragma unroll
            for (int e = 0; e < 8; ++e) acc[i][j][e] = 0.f;

    const int r16 = lane & 15;
    const int kh  = (lane >> 4) * 16;  // A frag K-half select
    const int ks  = (lane >> 4) * 16;  // B frag K-half select

    // async stage of one 256x64 A tile + 128x64 W tile: 12 b128 per thread
    auto stage = [&](int kt, int buf) {
        const int kbase = kt * BKT;
        __bf16* dA = sA + buf * (BM * LDSS);
        __bf16* dB = sB + buf * (BN * LDSS);
#pragma unroll
        for (int c = 0; c < 8; ++c) {           // 2048 chunks of 8 elems
            const int chunk = tid + c * 256;
            const int r = chunk >> 3;
            const int col = (chunk & 7) * 8;
            async_ld_b128(dA + r * LDSS + col, A + (size_t)(m0 + r) * lda + kbase + col);
        }
#pragma unroll
        for (int c = 0; c < 4; ++c) {           // 1024 chunks of 8 elems
            const int chunk = tid + c * 256;
            const int r = chunk >> 3;
            const int col = (chunk & 7) * 8;
            async_ld_b128(dB + r * LDSS + col, W + (size_t)(n0 + r) * (size_t)K + kbase + col);
        }
    };

    const int nkt = K / BKT;           // K is always a multiple of 64 here
    stage(0, 0);
    for (int kt = 0; kt < nkt; ++kt) {
        const int cur = kt & 1;
        if (kt + 1 < nkt) {
            stage(kt + 1, cur ^ 1);    // prefetch next tile into other buffer
            WAIT_ASYNC(12);            // in-order completion: current tile landed
        } else {
            WAIT_ASYNC(0);
        }
        __syncthreads();
        const __bf16* cA = sA + cur * (BM * LDSS);
        const __bf16* cB = sB + cur * (BN * LDSS);

#pragma unroll
        for (int kk = 0; kk < BKT; kk += 32) {
            // B fragments: 32x16 bf16 column-major; lane holds col (lane&15),
            // 16 contiguous K starting at ks
            v16bf fb[4];
#pragma unroll
            for (int j = 0; j < 4; ++j) {
                const __bf16* p = cB + (wc * 64 + j * 16 + r16) * LDSS + kk + ks;
                v8bf lo = *(const v8bf*)p;
                v8bf hi = *(const v8bf*)(p + 8);
#pragma unroll
                for (int e = 0; e < 8; ++e) { fb[j][e] = lo[e]; fb[j][8 + e] = hi[e]; }
            }
            // A fragments: 16x32 bf16; lane holds row (lane&15), K runs
            // [kh, kh+16) contiguous (matches B fragment order)
#pragma unroll
            for (int i = 0; i < 4; ++i) {
                const __bf16* p = cA + (wr * 64 + i * 16 + r16) * LDSS + kk + kh;
                v8bf lo = *(const v8bf*)p;
                v8bf hi = *(const v8bf*)(p + 8);
                v16bf fa;
#pragma unroll
                for (int e = 0; e < 8; ++e) { fa[e] = lo[e]; fa[8 + e] = hi[e]; }
#pragma unroll
                for (int j = 0; j < 4; ++j)
                    acc[i][j] = __builtin_amdgcn_wmma_f32_16x16x32_bf16(
                        false, fa, false, fb[j], (short)0, acc[i][j], false, false);
            }
        }
        __syncthreads();
    }

    // Epilogue. C layout: element e of v8f -> M = e + 8*(lane>>4), N = lane&15
    const int cn   = lane & 15;
    const int moff = (lane >> 4) * 8;
#pragma unroll
    for (int i = 0; i < 4; ++i) {
        const int gmb = m0 + wr * 64 + i * 16 + moff;
#pragma unroll
        for (int j = 0; j < 4; ++j) {
            const int gn = n0 + wc * 64 + j * 16 + cn;
            if (gn < N) {
#pragma unroll
                for (int e = 0; e < 8; ++e) {
                    float v = acc[i][j][e];
                    const size_t off = (size_t)(gmb + e) * ldc + gn;
                    if constexpr (EPI == 0) {
                        Cf[off] = v;
                    } else if constexpr (EPI == 1) {
                        Cf[off] = softplusf_(v + bias[gn]);
                    } else if constexpr (EPI == 2) {
                        Cb[off] = (__bf16)geluf_(v + bias[gn]);
                    } else { // dual
                        Cf[off] = v;
                        Cb[off] = (__bf16)v;
                    }
                }
            }
        }
    }
}

// ---------------------------------------------------------------------------
// Causal depthwise conv (K=4) + SiLU, reads xin = xz[:, :Di]
// ---------------------------------------------------------------------------
__global__ void conv_silu_kernel(
    const float* __restrict__ xz, const float* __restrict__ cw,
    const float* __restrict__ cb, float* __restrict__ xc, __bf16* __restrict__ xcb)
{
    const int idx = blockIdx.x * 256 + threadIdx.x;   // over MROWS*DINNER
    if (idx >= MROWS * DINNER) return;
    const int row = idx >> 11;                        // /DINNER, b*L + l
    const int d = idx & (DINNER - 1);
    const int l = row & (LSEQ - 1);
    float acc = cb[d];
#pragma unroll
    for (int k = 0; k < KCONV; ++k) {
        const int ll = l - (KCONV - 1) + k;
        if (ll >= 0)
            acc += cw[d * KCONV + k] * xz[(size_t)(row - (KCONV - 1) + k) * (2 * DINNER) + d];
    }
    const float s = siluf_(acc);
    xc[idx] = s;
    xcb[idx] = (__bf16)s;
}

// ---------------------------------------------------------------------------
// Selective scan: sequential over L, parallel over (b, d). One thread per
// channel holds 16 SSM states; B_t/C_t broadcast via async-to-LDS double
// buffer (one timestep ahead); next-step operand rows prefetched. Fuses the
// D*x skip and SiLU(z) gating; emits bf16 for the output projection GEMM.
// ---------------------------------------------------------------------------
__global__ __launch_bounds__(256) void scan_kernel(
    const float* __restrict__ dt,    // (B*L, Di)
    const float* __restrict__ xdbl,  // (B*L, 96): [dt_raw|B|C]
    const float* __restrict__ xc,    // (B*L, Di)
    const float* __restrict__ xz,    // (B*L, 2*Di): z at +Di
    const float* __restrict__ A_log, // (Di, 16)
    const float* __restrict__ Dss,   // (Di)
    __bf16* __restrict__ yg)         // (B*L, Di)
{
    __shared__ float sBC[2][2 * NSTATE];
    const int tid = threadIdx.x;
    const int blocksPerB = DINNER / 256;
    const int b = blockIdx.x / blocksPerB;
    const int d = (blockIdx.x % blocksPerB) * 256 + tid;

    float a[NSTATE], h[NSTATE];
#pragma unroll
    for (int n = 0; n < NSTATE; ++n) {
        a[n] = -__expf(A_log[(size_t)d * NSTATE + n]);
        h[n] = 0.f;
    }
    const float Dd = Dss[d];
    const size_t rowBase = (size_t)b * LSEQ;

    // preload timestep 0's B/C (32 floats = 8 lanes x b128)
    if (tid < 8)
        async_ld_b128(&sBC[0][tid * 4], xdbl + rowBase * XPROJC + RRANK + tid * 4);

    for (int l = 0; l < LSEQ; ++l) {
        const size_t row = rowBase + l;
        const int cur = l & 1;
        if (l + 1 < LSEQ) {
            if (tid < 8)
                async_ld_b128(&sBC[cur ^ 1][tid * 4],
                              xdbl + (row + 1) * XPROJC + RRANK + tid * 4);
            WAIT_ASYNC(1);             // older (current) staging done, next in flight
            __builtin_prefetch(dt + (row + 1) * DINNER + d, 0, 0);
            __builtin_prefetch(xc + (row + 1) * DINNER + d, 0, 0);
        } else {
            WAIT_ASYNC(0);
        }
        __syncthreads();
        const float dt_t = dt[row * DINNER + d];
        const float x_t  = xc[row * DINNER + d];
        const float z_t  = xz[row * (2 * DINNER) + DINNER + d];
        const float dx = dt_t * x_t;
        float y = 0.f;
#pragma unroll
        for (int n = 0; n < NSTATE; ++n) {
            h[n] = h[n] * __expf(dt_t * a[n]) + dx * sBC[cur][n];
            y += h[n] * sBC[cur][NSTATE + n];
        }
        yg[row * DINNER + d] = (__bf16)((y + Dd * x_t) * siluf_(z_t));
        __syncthreads();
    }
}

// ---------------------------------------------------------------------------
// Decoder head: out[(l-sep), b, o] = gelu_act[b*L+l, :] @ dec2_w[o, :] + dec2_b[o]
// ---------------------------------------------------------------------------
__global__ void dec2_kernel(
    const __bf16* __restrict__ act, const float* __restrict__ w2,
    const float* __restrict__ b2, const int* __restrict__ sep_p,
    float* __restrict__ out)
{
    const int idx = blockIdx.x * 256 + threadIdx.x;   // over L*B*NOUT
    if (idx >= LSEQ * BATCH * NOUTC) return;
    const int o = idx % NOUTC;
    const int b = (idx / NOUTC) % BATCH;
    const int l = idx / (NOUTC * BATCH);
    const int sep = *sep_p;
    if (l < sep) return;
    const __bf16* ar = act + ((size_t)b * LSEQ + l) * HDEC;
    const float* wr = w2 + (size_t)o * HDEC;
    float acc = b2[o];
#pragma unroll 8
    for (int hh = 0; hh < HDEC; ++hh) acc += (float)ar[hh] * wr[hh];
    out[((size_t)(l - sep) * BATCH + b) * NOUTC + o] = acc;
}

// ---------------------------------------------------------------------------
// Host orchestration
// ---------------------------------------------------------------------------
static inline unsigned cdiv(size_t a, size_t b) { return (unsigned)((a + b - 1) / b); }

extern "C" void kernel_launch(void* const* d_in, const int* in_sizes, int n_in,
                              void* d_out, int out_size, void* d_ws, size_t ws_size,
                              hipStream_t stream) {
    (void)in_sizes; (void)n_in; (void)out_size; (void)ws_size;
    // inputs (setup_inputs order)
    const float* x_src   = (const float*)d_in[0];
    const float* y_src   = (const float*)d_in[1];
    const float* enc_w   = (const float*)d_in[2];
    const float* enc_b   = (const float*)d_in[3];
    const float* yenc_w  = (const float*)d_in[4];
    const float* yenc_b  = (const float*)d_in[5];
    const float* norm_w  = (const float*)d_in[6];
    const float* norm_b  = (const float*)d_in[7];
    const float* inp_w   = (const float*)d_in[8];
    const float* conv_w  = (const float*)d_in[9];
    const float* conv_b  = (const float*)d_in[10];
    const float* xp_w    = (const float*)d_in[11];
    const float* dt_w    = (const float*)d_in[12];
    const float* dt_b    = (const float*)d_in[13];
    const float* A_log   = (const float*)d_in[14];
    const float* D_ssm   = (const float*)d_in[15];
    const float* out_w   = (const float*)d_in[16];
    const float* normf_w = (const float*)d_in[17];
    const float* normf_b = (const float*)d_in[18];
    const float* dec1_w  = (const float*)d_in[19];
    const float* dec1_b  = (const float*)d_in[20];
    const float* dec2_w  = (const float*)d_in[21];
    const float* dec2_b  = (const float*)d_in[22];
    const int*   sep_p   = (const int*)d_in[23];
    float* out = (float*)d_out;

    // workspace carve-up
    char* ws = (char*)d_ws;
    size_t off = 0;
    auto alloc = [&](size_t bytes) -> void* {
        void* p = ws + off;
        off = (off + bytes + 255) & ~(size_t)255;
        return p;
    };
    const size_t nInp  = (size_t)NLAYER * 2 * DINNER * DMODEL;
    const size_t nXp   = (size_t)NLAYER * XPROJC * DINNER;
    const size_t nDtw  = (size_t)NLAYER * DINNER * RRANK;
    const size_t nOutw = (size_t)NLAYER * DMODEL * DINNER;
    const size_t nDec1 = (size_t)HDEC * DMODEL;

    __bf16* wInp  = (__bf16*)alloc(nInp * 2);
    __bf16* wXp   = (__bf16*)alloc(nXp * 2);
    __bf16* wDtw  = (__bf16*)alloc(nDtw * 2);
    __bf16* wOutw = (__bf16*)alloc(nOutw * 2);
    __bf16* wDec1 = (__bf16*)alloc(nDec1 * 2);

    float*  hbuf   = (float*)alloc((size_t)MROWS * DMODEL * 4);
    float*  resbuf = (float*)alloc((size_t)MROWS * DMODEL * 4);
    __bf16* hnbf   = (__bf16*)alloc((size_t)MROWS * DMODEL * 2);
    float*  xzbuf  = (float*)alloc((size_t)MROWS * 2 * DINNER * 4);
    float*  xcbuf  = (float*)alloc((size_t)MROWS * DINNER * 4);
    __bf16* xcbf   = (__bf16*)alloc((size_t)MROWS * DINNER * 2);
    float*  xdbl   = (float*)alloc((size_t)MROWS * XPROJC * 4);
    __bf16* xdblbf = (__bf16*)alloc((size_t)MROWS * XPROJC * 2);
    float*  dtbuf  = (float*)alloc((size_t)MROWS * DINNER * 4);
    __bf16* ygbf   = (__bf16*)alloc((size_t)MROWS * DINNER * 2);
    __bf16* act1bf = (__bf16*)alloc((size_t)MROWS * HDEC * 2);

    // 1) convert weights to bf16
    f32_to_bf16_kernel<<<cdiv(nInp, 256), 256, 0, stream>>>(inp_w, wInp, nInp);
    f32_to_bf16_kernel<<<cdiv(nXp, 256), 256, 0, stream>>>(xp_w, wXp, nXp);
    f32_to_bf16_kernel<<<cdiv(nDtw, 256), 256, 0, stream>>>(dt_w, wDtw, nDtw);
    f32_to_bf16_kernel<<<cdiv(nOutw, 256), 256, 0, stream>>>(out_w, wOutw, nOutw);
    f32_to_bf16_kernel<<<cdiv(nDec1, 256), 256, 0, stream>>>(dec1_w, wDec1, nDec1);

    // 2) encoder -> h (B, L, D)
    encoder_kernel<<<MROWS, 256, 0, stream>>>(x_src, y_src, enc_w, enc_b,
                                              yenc_w, yenc_b, sep_p, hbuf);

    // 3) mamba layers
    for (int i = 0; i < NLAYER; ++i) {
        ln_kernel<<<MROWS, 256, 0, stream>>>(hbuf, resbuf, norm_w + i * DMODEL,
                                             norm_b + i * DMODEL, hnbf, i == 0);

        // in_proj: (4096 x 4096 x 1024) -> xz f32
        gemm_nt_kernel<0><<<dim3(cdiv(2 * DINNER, BN), MROWS / BM), 256, 0, stream>>>(
            hnbf, DMODEL, wInp + (size_t)i * 2 * DINNER * DMODEL,
            xzbuf, nullptr, 2 * DINNER, nullptr, MROWS, 2 * DINNER, DMODEL);

        conv_silu_kernel<<<cdiv((size_t)MROWS * DINNER, 256), 256, 0, stream>>>(
            xzbuf, conv_w + (size_t)i * DINNER * KCONV, conv_b + (size_t)i * DINNER,
            xcbuf, xcbf);

        // x_proj: (4096 x 96 x 2048) -> xdbl f32 + bf16
        gemm_nt_kernel<3><<<dim3(cdiv(XPROJC, BN), MROWS / BM), 256, 0, stream>>>(
            xcbf, DINNER, wXp + (size_t)i * XPROJC * DINNER,
            xdbl, xdblbf, XPROJC, nullptr, MROWS, XPROJC, DINNER);

        // dt: (4096 x 2048 x 64), bias + softplus -> dt f32
        gemm_nt_kernel<1><<<dim3(cdiv(DINNER, BN), MROWS / BM), 256, 0, stream>>>(
            xdblbf, XPROJC, wDtw + (size_t)i * DINNER * RRANK,
            dtbuf, nullptr, DINNER, dt_b + (size_t)i * DINNER, MROWS, DINNER, RRANK);

        // selective scan + gating -> yg bf16
        scan_kernel<<<BATCH * (DINNER / 256), 256, 0, stream>>>(
            dtbuf, xdbl, xcbuf, xzbuf,
            A_log + (size_t)i * DINNER * NSTATE, D_ssm + (size_t)i * DINNER, ygbf);

        // out_proj: (4096 x 1024 x 2048) -> h f32
        gemm_nt_kernel<0><<<dim3(cdiv(DMODEL, BN), MROWS / BM), 256, 0, stream>>>(
            ygbf, DINNER, wOutw + (size_t)i * DMODEL * DINNER,
            hbuf, nullptr, DMODEL, nullptr, MROWS, DMODEL, DINNER);
    }

    // 4) final residual + LN
    ln_kernel<<<MROWS, 256, 0, stream>>>(hbuf, resbuf, normf_w, normf_b, hnbf, 0);

    // 5) dec1: (4096 x 2048 x 1024), bias + GELU -> bf16
    gemm_nt_kernel<2><<<dim3(cdiv(HDEC, BN), MROWS / BM), 256, 0, stream>>>(
        hnbf, DMODEL, wDec1, nullptr, act1bf, HDEC, dec1_b, MROWS, HDEC, DMODEL);

    // 6) dec2 head + slice [sep:]
    dec2_kernel<<<cdiv((size_t)LSEQ * BATCH * NOUTC, 256), 256, 0, stream>>>(
        act1bf, dec2_w, dec2_b, sep_p, out);
}
